// ModelSageConv_86586540687486
// MI455X (gfx1250) — matrix operand order repaired and checked by reading, not compiled, not run on hardware
//
#include <hip/hip_runtime.h>
#include <math.h>

#define NFD   256        // hidden feature width
#define MPAD  50048      // 50000 padded to multiple of 128 (8 waves x 16 rows)
#define NNODE 50000
#define NEDGE 800000

typedef __attribute__((ext_vector_type(16))) _Float16 v16h;
typedef __attribute__((ext_vector_type(8)))  _Float16 v8h;
typedef __attribute__((ext_vector_type(8)))  float    v8f;
typedef __attribute__((ext_vector_type(4)))  float    v4f;

// ---------------------------------------------------------------------------
// f32 -> f16 convert with zero-fill of the padded tail
// ---------------------------------------------------------------------------
__global__ void cvt_f16_kernel(const float* __restrict__ src, _Float16* __restrict__ dst,
                               long nsrc, long ntot) {
  long i = (long)blockIdx.x * blockDim.x + threadIdx.x;
  if (i < ntot) dst[i] = (i < nsrc) ? (_Float16)src[i] : (_Float16)0.0f;
}

// vectorized zero: one b128 store per thread
__global__ void zero_f32_kernel(float* __restrict__ p, long n4) {
  long i = (long)blockIdx.x * blockDim.x + threadIdx.x;
  if (i < n4) ((v4f*)p)[i] = (v4f){0.0f, 0.0f, 0.0f, 0.0f};
}

// ---------------------------------------------------------------------------
// WMMA GEMM:  out[M x 256] = A[M x K] @ W^T (+ A2[M x K2] @ W2^T) + bias
// W is [256 x K] row-major (PyTorch layout) == exactly WMMA's B-fragment
// layout (B[k][n] = W[n][k]).  f16 inputs, f32 accumulate.
// Register blocking: each wave computes a 16x64 tile (4 accumulators),
// reusing each A fragment across 4 B fragments.  All 4 B fragments are
// fetched as one ds_load burst before the 4 WMMAs so a single s_wait_dscnt
// covers 4 back-to-back v_wmma on independent accumulators.
// Block: 256 threads = 8 waves = 128 rows; blockIdx.y covers 64 cols.
// Weight tile (64 rows x K) staged in LDS with +8-half padded stride.
// ---------------------------------------------------------------------------
__device__ __forceinline__ void stage_w64(const _Float16* __restrict__ W, int n0, int K,
                                          _Float16* lds, int SK, int tid) {
  int chunks = K >> 3;                        // K/8 v8h chunks per row
  for (int idx = tid; idx < 64 * chunks; idx += 256) {
    int r = idx / chunks, c = idx - r * chunks;
    *(v8h*)(lds + r * SK + c * 8) = *(const v8h*)(W + (size_t)(n0 + r) * K + c * 8);
  }
}

__device__ __forceinline__ void wmma_loop4(const _Float16* __restrict__ A, int lda,
                                           const _Float16* lds, int SK, int K,
                                           int m0, int lane, v8f* acc) {
  const int row = lane & 15;                  // A: M index
  const int sel = lane >> 4;                  // half-select per ISA fragment layout
  const _Float16* pa = A + (size_t)(m0 + row) * lda;
  const _Float16* pb = lds + (lane & 15) * SK + sel * 8;
#pragma unroll 2
  for (int k0 = 0; k0 < K; k0 += 32) {
    // A 16x32 f16 fragment: lane<16 holds K 0..7 & 16..23; lane>=16 holds 8..15 & 24..31
    v8h alo = *(const v8h*)(pa + k0 + sel * 8);
    v8h ahi = *(const v8h*)(pa + k0 + 16 + sel * 8);
    // batch-fetch all 4 B fragments (8 ds_load_b128) before issuing any WMMA
    v8h blo[4], bhi[4];
#pragma unroll
    for (int j = 0; j < 4; ++j) {
      const _Float16* pbj = pb + (size_t)j * 16 * SK + k0;
      blo[j] = *(const v8h*)pbj;
      bhi[j] = *(const v8h*)(pbj + 16);
    }
    v16h a;
#pragma unroll
    for (int i = 0; i < 8; ++i) { a[i] = alo[i]; a[i + 8] = ahi[i]; }
#pragma unroll
    for (int j = 0; j < 4; ++j) {
      v16h b;
#pragma unroll
      for (int i = 0; i < 8; ++i) { b[i] = blo[j][i]; b[i + 8] = bhi[j][i]; }
      acc[j] = __builtin_amdgcn_wmma_f32_16x16x32_f16(false, a, false, b,
                                                      (short)0, acc[j], false, false);
    }
  }
}

__global__ __launch_bounds__(256) void gemm_wmma_kernel(
    const _Float16* __restrict__ A,  const _Float16* __restrict__ W,
    const float* __restrict__ bias,
    const _Float16* __restrict__ A2, const _Float16* __restrict__ W2,
    int K, int K2,
    float* __restrict__ outF, _Float16* __restrict__ outH, int relu) {
  __shared__ _Float16 ldsW[64 * 264];         // 64 rows x (K<=256 + 8 pad) halves
  const int tid  = threadIdx.x;
  const int n0   = blockIdx.y * 64;
  const int wave = tid >> 5, lane = tid & 31;
  const int m0   = blockIdx.x * 128 + wave * 16;

  v8f acc[4] = {{}, {}, {}, {}};

  // pass 1: A @ W^T
  stage_w64(W, n0, K, ldsW, K + 8, tid);
  __syncthreads();
  wmma_loop4(A, K, ldsW, K + 8, K, m0, lane, acc);

  // pass 2 (optional, fused): + A2 @ W2^T, re-staging same LDS buffer
  if (A2) {
    __syncthreads();                          // all reads of ldsW complete
    stage_w64(W2, n0, K2, ldsW, K2 + 8, tid);
    __syncthreads();
    wmma_loop4(A2, K2, ldsW, K2 + 8, K2, m0, lane, acc);
  }

  // C/D layout: VGPR g -> row m0 + 8*(lane>>4) + g, col n0 + 16j + lane%16
  const int col = lane & 15, sel = lane >> 4;
#pragma unroll
  for (int j = 0; j < 4; ++j) {
    const float bv = bias ? bias[n0 + j * 16 + col] : 0.0f;
    size_t base = (size_t)(m0 + sel * 8) * NFD + n0 + j * 16 + col;
#pragma unroll
    for (int g = 0; g < 8; ++g) {
      float v = acc[j][g] + bv;
      if (relu) v = fmaxf(v, 0.0f);
      if (outF) outF[base + (size_t)g * NFD] = v;
      if (outH) outH[base + (size_t)g * NFD] = (_Float16)v;
    }
  }
}

// ---------------------------------------------------------------------------
// Edge scatter: one wave per edge; 8 f32 atomics per lane into L2-resident acc
// ---------------------------------------------------------------------------
__global__ __launch_bounds__(256) void scatter_kernel(
    const int* __restrict__ srcI, const int* __restrict__ dstI, int E,
    const _Float16* __restrict__ H, float* __restrict__ acc, float* __restrict__ cnt) {
  int e = blockIdx.x * 8 + (threadIdx.x >> 5);
  if (e >= E) return;
  int lane = threadIdx.x & 31;
  int s = srcI[e], d = dstI[e];
  v8h h = *(const v8h*)(H + (size_t)s * NFD + lane * 8);
  float* a = acc + (size_t)d * NFD + lane * 8;
#pragma unroll
  for (int i = 0; i < 8; ++i) atomicAdd(a + i, (float)h[i]);
  if (lane == 0) atomicAdd(cnt + d, 1.0f);
}

__global__ void mean_kernel(const float* __restrict__ acc, const float* __restrict__ cnt,
                            _Float16* __restrict__ mh, long ntot) {
  long i = (long)blockIdx.x * blockDim.x + threadIdx.x;
  if (i < ntot) mh[i] = (_Float16)(acc[i] / fmaxf(cnt[i >> 8], 1.0f));
}

// Row L2 normalize: one wave per 256-float row, shuffle reduction, f16 out
__global__ __launch_bounds__(256) void normalize_kernel(
    const float* __restrict__ in, _Float16* __restrict__ out, int rows) {
  int row = blockIdx.x * 8 + (threadIdx.x >> 5);
  if (row >= rows) return;
  int lane = threadIdx.x & 31;
  const float* p = in + (size_t)row * NFD + lane * 8;
  float v[8], ss = 0.0f;
#pragma unroll
  for (int i = 0; i < 8; ++i) { v[i] = p[i]; ss += v[i] * v[i]; }
  for (int m = 16; m >= 1; m >>= 1) ss += __shfl_xor(ss, m, 32);
  float s = 1.0f / fmaxf(sqrtf(ss), 1e-12f);
  _Float16* q = out + (size_t)row * NFD + lane * 8;
#pragma unroll
  for (int i = 0; i < 8; ++i) q[i] = (_Float16)(v[i] * s);
}

// Output head: [M x 256] f16 @ Wo2^T[256 x 2] + bo2 -> f32, one wave per row
__global__ __launch_bounds__(256) void head_kernel(
    const _Float16* __restrict__ h, const float* __restrict__ Wo2,
    const float* __restrict__ bo2, float* __restrict__ out, int M) {
  int row = blockIdx.x * 8 + (threadIdx.x >> 5);
  if (row >= M) return;
  int lane = threadIdx.x & 31;
  const _Float16* p = h + (size_t)row * NFD + lane * 8;
  float s0 = 0.0f, s1 = 0.0f;
#pragma unroll
  for (int i = 0; i < 8; ++i) {
    float hv = (float)p[i];
    int k = lane * 8 + i;
    s0 += hv * Wo2[k];
    s1 += hv * Wo2[NFD + k];
  }
  for (int m = 16; m >= 1; m >>= 1) { s0 += __shfl_xor(s0, m, 32); s1 += __shfl_xor(s1, m, 32); }
  if (lane == 0) {
    out[(size_t)row * 2]     = s0 + bo2[0];
    out[(size_t)row * 2 + 1] = s1 + bo2[1];
  }
}

// ---------------------------------------------------------------------------
// Host driver
// ---------------------------------------------------------------------------
extern "C" void kernel_launch(void* const* d_in, const int* in_sizes, int n_in,
                              void* d_out, int out_size, void* d_ws, size_t ws_size,
                              hipStream_t stream) {
  (void)in_sizes; (void)n_in; (void)out_size; (void)ws_size;
  const float* x_rec = (const float*)d_in[0];
  const float* x_con = (const float*)d_in[1];
  const int*   e_rc  = (const int*)d_in[2];   // [0]=rec src, [1]=con dst
  const int*   e_cr  = (const int*)d_in[3];   // [0]=con src, [1]=rec dst
  const float* Wi_rec = (const float*)d_in[4];  const float* bi_rec = (const float*)d_in[5];
  const float* Wi_con = (const float*)d_in[6];  const float* bi_con = (const float*)d_in[7];
  // layer param index blocks: Wp, bp, Wl, bl, Wr  for rc1, cr1, rc2, cr2
  const int LW[4] = {8, 13, 18, 23};
  const float* Wo1 = (const float*)d_in[28]; const float* bo1 = (const float*)d_in[29];
  const float* Wo2 = (const float*)d_in[30]; const float* bo2 = (const float*)d_in[31];

  // -------- workspace carving --------
  char* wsp = (char*)d_ws;
  size_t off = 0;
  auto carve = [&](size_t bytes) -> void* {
    void* p = wsp + off;
    off = (off + bytes + 255) & ~(size_t)255;
    return p;
  };
  _Float16* xrec_h = (_Float16*)carve((size_t)MPAD * 64 * 2);
  _Float16* xcon_h = (_Float16*)carve((size_t)MPAD * 32 * 2);
  _Float16* r_h    = (_Float16*)carve((size_t)MPAD * NFD * 2);
  _Float16* c_h    = (_Float16*)carve((size_t)MPAD * NFD * 2);
  _Float16* Hbuf   = (_Float16*)carve((size_t)MPAD * NFD * 2);  // messages, then mean
  float*    acc    = (float*)   carve((size_t)MPAD * NFD * 4);  // scatter acc, then pre-norm out
  float*    cnt    = (float*)   carve((size_t)MPAD * 4);
  _Float16* wi_rec_h = (_Float16*)carve(256 * 64 * 2);
  _Float16* wi_con_h = (_Float16*)carve(256 * 32 * 2);
  _Float16* wp_h[4], *wl_h[4], *wr_h[4];
  for (int l = 0; l < 4; ++l) {
    wp_h[l] = (_Float16*)carve(256 * 256 * 2);
    wl_h[l] = (_Float16*)carve(256 * 256 * 2);
    wr_h[l] = (_Float16*)carve(256 * 256 * 2);
  }
  _Float16* wo1_h = (_Float16*)carve(256 * 256 * 2);

  auto cvt = [&](const float* s, _Float16* d, long nsrc, long ntot) {
    hipLaunchKernelGGL(cvt_f16_kernel, dim3((unsigned)((ntot + 255) / 256)), dim3(256),
                       0, stream, s, d, nsrc, ntot);
  };
  auto zero = [&](float* p, long n) {   // n must be a multiple of 4
    long n4 = n >> 2;
    hipLaunchKernelGGL(zero_f32_kernel, dim3((unsigned)((n4 + 255) / 256)), dim3(256),
                       0, stream, p, n4);
  };
  auto gemm = [&](const _Float16* A, const _Float16* W, const float* bias,
                  const _Float16* A2, const _Float16* W2, int K, int K2,
                  float* oF, _Float16* oH, int relu) {
    hipLaunchKernelGGL(gemm_wmma_kernel, dim3(MPAD / 128, NFD / 64), dim3(256),
                       0, stream, A, W, bias, A2, W2, K, K2, oF, oH, relu);
  };

  // -------- f16 conversions (inputs padded with zeros, weights exact) --------
  cvt(x_rec, xrec_h, (long)NNODE * 64, (long)MPAD * 64);
  cvt(x_con, xcon_h, (long)NNODE * 32, (long)MPAD * 32);
  cvt(Wi_rec, wi_rec_h, 256 * 64, 256 * 64);
  cvt(Wi_con, wi_con_h, 256 * 32, 256 * 32);
  for (int l = 0; l < 4; ++l) {
    cvt((const float*)d_in[LW[l] + 0], wp_h[l], 256 * 256, 256 * 256);
    cvt((const float*)d_in[LW[l] + 2], wl_h[l], 256 * 256, 256 * 256);
    cvt((const float*)d_in[LW[l] + 4], wr_h[l], 256 * 256, 256 * 256);
  }
  cvt(Wo1, wo1_h, 256 * 256, 256 * 256);

  // -------- input projections (no relu) --------
  gemm(xrec_h, wi_rec_h, bi_rec, nullptr, nullptr, 64, 0, nullptr, r_h, 0);
  gemm(xcon_h, wi_con_h, bi_con, nullptr, nullptr, 32, 0, nullptr, c_h, 0);

  // -------- SAGE layer driver --------
  auto sage = [&](const _Float16* src, _Float16* dst, const int* edges, int l) {
    const float* bp = (const float*)d_in[LW[l] + 1];
    const float* bl = (const float*)d_in[LW[l] + 3];
    // H = relu(src @ Wp^T + bp)
    gemm(src, wp_h[l], bp, nullptr, nullptr, NFD, 0, nullptr, Hbuf, 1);
    zero(acc, (long)MPAD * NFD);
    zero(cnt, MPAD);
    hipLaunchKernelGGL(scatter_kernel, dim3(NEDGE / 8), dim3(256), 0, stream,
                       edges, edges + NEDGE, NEDGE, Hbuf, acc, cnt);
    hipLaunchKernelGGL(mean_kernel, dim3(MPAD * NFD / 256), dim3(256), 0, stream,
                       acc, cnt, Hbuf, (long)MPAD * NFD);   // Hbuf := mean (f16)
    // out = mean @ Wl^T + bl + dst @ Wr^T  (fused dual-WMMA, f32 out in acc)
    gemm(Hbuf, wl_h[l], bl, dst, wr_h[l], NFD, NFD, acc, nullptr, 0);
    hipLaunchKernelGGL(normalize_kernel, dim3(MPAD / 8), dim3(256), 0, stream,
                       acc, dst, MPAD);                     // dst := normalized f16
  };

  sage(r_h, c_h, e_rc, 0);   // rc1
  sage(c_h, r_h, e_cr, 1);   // cr1
  sage(r_h, c_h, e_rc, 2);   // rc2
  sage(c_h, r_h, e_cr, 3);   // cr2

  // -------- output head --------
  gemm(r_h, wo1_h, bo1, nullptr, nullptr, NFD, 0, nullptr, Hbuf, 1);
  hipLaunchKernelGGL(head_kernel, dim3((NNODE + 7) / 8), dim3(256), 0, stream,
                     Hbuf, Wo2, bo2, (float*)d_out, NNODE);
}
